// PytorchFlexAttention_19335942767119
// MI455X (gfx1250) — compile-verified, matrix operand order checked by reading, not run on hardware
//
#include <hip/hip_runtime.h>
#include <math.h>

// ---------------------------------------------------------------------------
// Fused GQA flash-attention forward for gfx1250 (MI455X), f16 WMMA, f32 accum.
// S=2048, B=2, Hq=32, Hkv=8, D=128, dense bool (causal) mask.
//
// Per block: 8 waves x 16 query rows = 128 q-rows of one (b,h).
// Key tiles of 32 double-buffered in LDS (global loads for tile i+1 issued
// before tile i's WMMAs; one barrier/iter). Per tile per wave: 8 QK^T WMMAs
// (K-fragments preloaded so ds waits stagger off the WMMA chain), online
// softmax in exp2 domain (raw v_exp_f32; masked entries underflow to 0 via
// NEG_BIG so the update block is branch-free), 1 row-sum WMMA (P x ones),
// 8 software-pipelined PV WMMAs.
// ---------------------------------------------------------------------------

typedef __attribute__((ext_vector_type(16))) _Float16 v16h;
typedef __attribute__((ext_vector_type(8)))  _Float16 v8h;
typedef __attribute__((ext_vector_type(8)))  float    v8f;
typedef __attribute__((ext_vector_type(4)))  float    f32x4;

#define S_LEN   2048
#define BATCH   2
#define HQ      32
#define HKV     8
#define DH      128
#define GQA_REP (HQ / HKV)
// (1/sqrt(128)) * log2(e): softmax in exp2 domain -> raw v_exp_f32
#define SCALE_LOG2E 0.12751739348898342f
#define NEG_BIG (-3.0e38f)

#define KROW 136   // Ksh row pitch (halves): 272B -> bank stride 4/lane, conflict-free
#define PROW 40    // Psh row pitch (halves): 80B  -> conflict-free pa loads

// Fragment maps (ISA 7.12.2, 16-bit data, wave32, lane = 16*g + ln):
//   A/B frag v16h element e -> K = (e<8 ? g*8+e : 16 + g*8 + (e-8))
//   C/D f32 16x16: VGPR r   -> (M = r + 8*g, N = ln)

__global__ __launch_bounds__(256)
void flash_attn_fwd(const float* __restrict__ Q,
                    const float* __restrict__ K,
                    const float* __restrict__ V,
                    const unsigned char* __restrict__ MASK,
                    float* __restrict__ OUT)
{
    __shared__ __align__(16) _Float16 Ksh[2][32][KROW];      // [buf][key][d]  17.4 KB
    __shared__ __align__(16) _Float16 Vsh[2][DH * 32];       // d-major swizzled 16 KB
    __shared__ __align__(16) _Float16 Psh[8 * 16 * PROW];    // per-wave P      10 KB

    const int tid  = threadIdx.x;
    const int wave = tid >> 5;
    const int lane = tid & 31;
    const int g    = lane >> 4;
    const int ln   = lane & 15;

    const int b   = blockIdx.y;
    const int h   = blockIdx.z;
    const int hkv = h / GQA_REP;
    const int qb  = blockIdx.x * 128 + wave * 16;

    // ---- Q tile as 4 A-fragments (16x32 f16), loaded once ------------------
    const float* qrow = Q + (((size_t)(qb + ln) * BATCH + b) * HQ + h) * DH;
    v16h qa[4];
    #pragma unroll
    for (int c = 0; c < 4; ++c) {
        const float* p0 = qrow + c * 32 + g * 8;        // K = g*8 + e
        const float* p1 = p0 + 16;                      // K = 16 + g*8 + e
        #pragma unroll
        for (int e = 0; e < 8; ++e) {
            qa[c][e]     = (_Float16)p0[e];
            qa[c][8 + e] = (_Float16)p1[e];
        }
    }

    // ---- constant all-ones B fragment: row-sum via the matrix pipe ---------
    v16h ones;
    #pragma unroll
    for (int e = 0; e < 16; ++e) ones[e] = (_Float16)1.0f;

    // ---- online-softmax state + O accumulators (16x128 f32) ----------------
    float rowm[8], rowl[8];
    v8f   o[8];
    {
        v8f z = {};
        #pragma unroll
        for (int f = 0; f < 8; ++f) o[f] = z;
        #pragma unroll
        for (int r = 0; r < 8; ++r) { rowm[r] = NEG_BIG; rowl[r] = 0.0f; }
    }

    // ---- staging slice for this thread (tile shared by whole block) --------
    const int skey = tid >> 3;            // 0..31
    const int sd0  = (tid & 7) * 16;      // 16 floats
    const size_t kvRow = (size_t)BATCH * HKV * DH;       // floats per key row
    const float* kp = K + ((size_t)skey * BATCH + b) * (HKV * DH) + hkv * DH + sd0;
    const float* vp = V + ((size_t)skey * BATCH + b) * (HKV * DH) + hkv * DH + sd0;
    const int vswzR = (ln >> 2) & 3;      // read-side V segment swizzle

    f32x4 kq[4], vq[4];                   // in-flight staging registers

    auto tile_load = [&](int kb) {        // issue 8 x global_load_b128
        const float* kt = kp + (size_t)kb * kvRow;
        const float* vt = vp + (size_t)kb * kvRow;
        #pragma unroll
        for (int j = 0; j < 4; ++j) {
            kq[j] = ((const f32x4*)kt)[j];
            vq[j] = ((const f32x4*)vt)[j];
        }
    };
    auto tile_store = [&](int buf) {      // f32 -> f16 convert + LDS store
        v8h h0, h1;
        #pragma unroll
        for (int e = 0; e < 8; ++e) {
            h0[e] = (_Float16)kq[e >> 2][e & 3];
            h1[e] = (_Float16)kq[2 + (e >> 2)][e & 3];
        }
        *(v8h*)&Ksh[buf][skey][sd0]     = h0;
        *(v8h*)&Ksh[buf][skey][sd0 + 8] = h1;
        // V: d-major, 8-half segments XOR-swizzled by (d>>2)&3
        #pragma unroll
        for (int i = 0; i < 16; ++i) {
            const int d   = sd0 + i;
            const int seg = (skey >> 3) ^ ((i >> 2) & 3);
            Vsh[buf][d * 32 + seg * 8 + (skey & 7)] = (_Float16)vq[i >> 2][i & 3];
        }
    };

    // ---- pipeline prologue: stage tile 0 -----------------------------------
    tile_load(0);
    tile_store(0);
    __syncthreads();
    int cur = 0;

    // =======================================================================
    for (int kb = 0; kb < S_LEN; kb += 32) {
        const bool have_next = (kb + 32 < S_LEN);
        if (have_next) tile_load(kb + 32);     // loads in flight during WMMAs

        // ---- mask bits for this wave's 16x32 score tile -------------------
        unsigned mbits[2];
        #pragma unroll
        for (int t = 0; t < 2; ++t) {
            unsigned bits = 0;
            const size_t kcol = (size_t)kb + 16 * t + ln;
            #pragma unroll
            for (int r = 0; r < 8; ++r) {
                const size_t qi = (size_t)qb + r + 8 * g;
                if (MASK[qi * S_LEN + kcol]) bits |= (1u << r);
            }
            mbits[t] = bits;
        }

        if (__any((mbits[0] | mbits[1]) != 0)) {   // causal: skips ~half the tiles

            // ---- scores: K-fragments preloaded, then 4 chained WMMAs ------
            v8f s[2];
            #pragma unroll
            for (int t = 0; t < 2; ++t) {
                const int key = 16 * t + ln;             // B-frag: N = key
                v16h kf[4];
                #pragma unroll
                for (int c = 0; c < 4; ++c) {            // 8 ds_load_b128, batched
                    const v8h lo = *(const v8h*)&Ksh[cur][key][c * 32 + g * 8];
                    const v8h hi = *(const v8h*)&Ksh[cur][key][c * 32 + 16 + g * 8];
                    kf[c] = __builtin_shufflevector(lo, hi,
                        0,1,2,3,4,5,6,7,8,9,10,11,12,13,14,15);
                }
                v8f acc = {};
                #pragma unroll
                for (int c = 0; c < 4; ++c)
                    acc = __builtin_amdgcn_wmma_f32_16x16x32_f16(
                        false, qa[c], false, kf[c], (short)0, acc, false, false);
                s[t] = acc;
            }

            // ---- scale (+ mask only for partial tiles) + per-row max ------
            // Masked entries become NEG_BIG; downstream exp2 underflows to 0,
            // so everything after this step is branch-free and shared.
            const bool full = __all((mbits[0] & mbits[1]) == 0xFFu);
            float tmax[8];
            if (full) {
                #pragma unroll
                for (int r = 0; r < 8; ++r) {
                    s[0][r] *= SCALE_LOG2E;
                    s[1][r] *= SCALE_LOG2E;
                    tmax[r] = fmaxf(s[0][r], s[1][r]);
                }
            } else {
                #pragma unroll
                for (int r = 0; r < 8; ++r) {
                    s[0][r] = ((mbits[0] >> r) & 1) ? s[0][r] * SCALE_LOG2E : NEG_BIG;
                    s[1][r] = ((mbits[1] >> r) & 1) ? s[1][r] * SCALE_LOG2E : NEG_BIG;
                    tmax[r] = fmaxf(s[0][r], s[1][r]);
                }
            }
            #pragma unroll
            for (int off = 1; off < 16; off <<= 1) {
                #pragma unroll
                for (int r = 0; r < 8; ++r)
                    tmax[r] = fmaxf(tmax[r], __shfl_xor(tmax[r], off, 32));
            }

            // ---- unified online-softmax update (raw v_exp_f32) ------------
            const int pbase = wave * 16 * PROW;
            #pragma unroll
            for (int r = 0; r < 8; ++r) {
                const float mnew  = fmaxf(rowm[r], tmax[r]);
                const float alpha = __builtin_amdgcn_exp2f(rowm[r] - mnew);
                const float p0    = __builtin_amdgcn_exp2f(s[0][r] - mnew);
                const float p1    = __builtin_amdgcn_exp2f(s[1][r] - mnew);
                rowm[r] = mnew;
                rowl[r] *= alpha;
                #pragma unroll
                for (int f = 0; f < 8; ++f) o[f][r] *= alpha;   // co-executes w/ WMMA
                Psh[pbase + (r + 8 * g) * PROW + ln]      = (_Float16)p0;
                Psh[pbase + (r + 8 * g) * PROW + 16 + ln] = (_Float16)p1;
            }

            // ---- re-read P in A-fragment layout (LDS in-order per wave) ---
            v16h pa;
            {
                const _Float16* pr = &Psh[pbase + ln * PROW];
                const v8h lo = *(const v8h*)&pr[g * 8];
                const v8h hi = *(const v8h*)&pr[16 + g * 8];
                pa = __builtin_shufflevector(lo, hi,
                    0,1,2,3,4,5,6,7,8,9,10,11,12,13,14,15);
            }

            // ---- row-sum of P on the matrix pipe: P(16x32) x ones(32x16) --
            {
                v8f lsum = {};
                lsum = __builtin_amdgcn_wmma_f32_16x16x32_f16(
                    false, pa, false, ones, (short)0, lsum, false, false);
                #pragma unroll
                for (int r = 0; r < 8; ++r) rowl[r] += lsum[r];
            }

            // ---- PV: 8 WMMAs, V-fragments pipelined by one ----------------
            auto vfrag = [&](int f) -> v16h {
                const int d = f * 16 + ln;               // B-frag: N = d column
                const v8h lo = *(const v8h*)&Vsh[cur][d * 32 + ((0 + g) ^ vswzR) * 8];
                const v8h hi = *(const v8h*)&Vsh[cur][d * 32 + ((2 + g) ^ vswzR) * 8];
                return __builtin_shufflevector(lo, hi,
                    0,1,2,3,4,5,6,7,8,9,10,11,12,13,14,15);
            };
            v16h vcurf = vfrag(0);
            #pragma unroll
            for (int f = 0; f < 8; ++f) {
                v16h vnext = vcurf;
                if (f < 7) vnext = vfrag(f + 1);         // loads under this WMMA
                o[f] = __builtin_amdgcn_wmma_f32_16x16x32_f16(
                    false, pa, false, vcurf, (short)0, o[f], false, false);
                vcurf = vnext;
            }
        }

        // ---- stage next tile into the other buffer; one barrier/iter ------
        if (have_next) tile_store(cur ^ 1);
        __syncthreads();
        cur ^= 1;
    }

    // ---- epilogue: normalize and store fp32 output [S, B, Hq*D] -----------
    float inv[8];
    #pragma unroll
    for (int r = 0; r < 8; ++r)
        inv[r] = (rowl[r] > 0.0f) ? (1.0f / rowl[r]) : 0.0f;

    #pragma unroll
    for (int f = 0; f < 8; ++f) {
        #pragma unroll
        for (int r = 0; r < 8; ++r) {
            const size_t qi = (size_t)qb + r + 8 * g;
            OUT[((qi * BATCH + b) * HQ + h) * DH + f * 16 + ln] = o[f][r] * inv[r];
        }
    }
}

// ---------------------------------------------------------------------------
extern "C" void kernel_launch(void* const* d_in, const int* in_sizes, int n_in,
                              void* d_out, int out_size, void* d_ws, size_t ws_size,
                              hipStream_t stream)
{
    const float* q = (const float*)d_in[0];
    const float* k = (const float*)d_in[1];
    const float* v = (const float*)d_in[2];
    const unsigned char* mask = (const unsigned char*)d_in[3];  // bool = 1 byte
    float* out = (float*)d_out;

    dim3 grid(S_LEN / 128, BATCH, HQ);   // (16, 2, 32)
    flash_attn_fwd<<<grid, 256, 0, stream>>>(q, k, v, mask, out);
}
